// Net_62929860821324
// MI455X (gfx1250) — matrix operand order repaired and checked by reading, not compiled, not run on hardware
//
#include <hip/hip_runtime.h>
#include <hip/hip_bf16.h>

// ---------------------------------------------------------------------------
// INRF-CNN forward for MI455X (gfx1250, wave32, WMMA).
// Implicit-GEMM with v_wmma_f32_16x16x32_f16; comb() computed on the fly.
// Block computes full Cout=192 x 64(N) tile so each comb/atan is computed
// exactly once (atan:MAC = 1:192). Layer-2/3 inputs staged NHWC so comb
// loads are 128-bit vectors.
// ---------------------------------------------------------------------------

typedef __attribute__((ext_vector_type(16))) _Float16 v16h;
typedef __attribute__((ext_vector_type(8)))  float    v8f;

union Frag16 { v16h v; uint4 q[2]; };
union Pack8h { _Float16 h[8]; uint4 q; };

#define LAM  2.0f
#define PSIG 12.0f
#define BN_EPS 1e-5f

// ---------------------------------------------------------------------------
// Weight packing: w[o][i][dy][dx] (f32) -> wpk[o*Kpad + (q*Cin + i)] (f16),
// zero padded for k >= 25*Cin.
// ---------------------------------------------------------------------------
__global__ void pack_weights(const float* __restrict__ w, _Float16* __restrict__ wpk,
                             int Cin, int Kpad) {
    int idx = blockIdx.x * 256 + threadIdx.x;
    int total = 192 * Kpad;
    if (idx >= total) return;
    int o = idx / Kpad;
    int k = idx - o * Kpad;
    int K = Cin * 25;
    float v = 0.0f;
    if (k < K) {
        int q = k / Cin;
        int i = k - q * Cin;
        v = w[(o * Cin + i) * 25 + q];   // q = dy*5+dx, contiguous in w
    }
    wpk[idx] = (_Float16)v;
}

// ---------------------------------------------------------------------------
// Implicit-GEMM INRF layer, M=192 (all Cout) x N=64 tile per block.
// XNHWC: x is [B, H*W, CIN] (vectorized comb loads); else x is [B, CIN, H, W].
// 256 thr = 8 waves in 4(M) x 2(N); wave tile = 48(M) x 32(N) = 6 WMMAs/step.
// Output always NCHW [B,192,H,W] with fused +bias and relu.
// ---------------------------------------------------------------------------
template<int CIN, int H, int W, int KPAD, int KREAL, bool XNHWC>
__global__ __launch_bounds__(256)
void inrf_gemm(const float* __restrict__ x, const _Float16* __restrict__ wpk,
               const float* __restrict__ bias, float* __restrict__ out) {
    constexpr int HW = H * W;
    constexpr int LDSTR = 40;                  // halfs per row (80B, bank spread)
    __shared__ _Float16 sA[192 * LDSTR];       // [m][k]  weights tile (192x32)
    __shared__ _Float16 sB[64 * LDSTR];        // [n][k]  comb tile    (64x32)

    const int tid  = threadIdx.x;
    const int n0   = blockIdx.x * 64;
    const int lane = tid & 31;
    const int wave = tid >> 5;
    const int wm   = wave >> 1;                // 0..3 -> M sub-tile of 48
    const int wn   = wave & 1;                 // 0..1 -> N sub-tile of 32

    // staging assignment: row = tid>>2 (0..63), k-segment of 8 = (tid&3)*8
    const int row = tid >> 2;
    const int ks  = (tid & 3) * 8;

    // decode this thread's comb output position (n = n0 + row)
    const int n  = n0 + row;
    const int b  = n / HW;
    const int hw = n - b * HW;
    const int h  = hw / W;
    const int w  = hw - h * W;

    v8f acc[3][2] = {};                        // [m-frag 0..2][n-frag 0..1]

    for (int k0 = 0; k0 < KPAD; k0 += 32) {
        __syncthreads();
        // ---- stage A: weights global -> LDS (3 x 16B per thread) ----
        #pragma unroll
        for (int rr = 0; rr < 3; ++rr) {
            const int m = rr * 64 + row;
            const uint4 wv = *(const uint4*)(wpk + (size_t)m * KPAD + k0 + ks);
            *(uint4*)(&sA[m * LDSTR + ks]) = wv;
            if (k0 + 32 < KPAD)   // prefetch next chunk (global_prefetch_b8)
                __builtin_prefetch(wpk + (size_t)m * KPAD + k0 + 32 + ks, 0, 1);
        }
        // ---- stage B: compute comb (8 elements per thread, 1 b128 store) ----
        Pack8h pk;
        if (CIN % 32 == 0) {
            // chunk never crosses a q boundary: hoist all index math
            const int kk0 = k0 + ks;
            const int q   = kk0 / CIN;
            const int i0  = kk0 - q * CIN;
            const int dy  = q / 5;
            const int dx  = q - dy * 5;
            const int hh  = h + dy - 2;
            const int ww  = w + dx - 2;
            const bool valid = (hh >= 0 && hh < H && ww >= 0 && ww < W);
            float ctr[8], pat[8];
            if (XNHWC) {
                const float* cp = x + ((size_t)b * HW + hw) * CIN + i0;
                const float* pp = x + ((size_t)b * HW + hh * W + ww) * CIN + i0;
                *(float4*)&ctr[0] = *(const float4*)cp;
                *(float4*)&ctr[4] = *(const float4*)(cp + 4);
                float4 p0 = {0.f, 0.f, 0.f, 0.f}, p1 = {0.f, 0.f, 0.f, 0.f};
                if (valid) { p0 = *(const float4*)pp; p1 = *(const float4*)(pp + 4); }
                *(float4*)&pat[0] = p0;
                *(float4*)&pat[4] = p1;
            } else {
                const float* cp = x + ((size_t)b * CIN + i0) * HW + hw;
                const float* pp = x + ((size_t)b * CIN + i0) * HW + hh * W + ww;
                #pragma unroll
                for (int j = 0; j < 8; ++j) {
                    ctr[j] = cp[j * HW];
                    pat[j] = valid ? pp[j * HW] : 0.0f;
                }
            }
            #pragma unroll
            for (int j = 0; j < 8; ++j) {
                const float comb = pat[j] - LAM * atanf(PSIG * (pat[j] - ctr[j]));
                pk.h[j] = (_Float16)comb;
            }
        } else {
            const float* xb = x + (size_t)b * CIN * HW;   // NCHW generic path
            #pragma unroll
            for (int j = 0; j < 8; ++j) {
                const int kk = k0 + ks + j;
                float comb = 0.0f;
                if (kk < KREAL) {
                    const int q  = kk / CIN;
                    const int i  = kk - q * CIN;
                    const int dy = q / 5;
                    const int dx = q - dy * 5;
                    const int hh = h + dy - 2;
                    const int ww = w + dx - 2;
                    const float center = xb[i * HW + hw];
                    float patch = 0.0f;
                    if (hh >= 0 && hh < H && ww >= 0 && ww < W)
                        patch = xb[i * HW + hh * W + ww];
                    comb = patch - LAM * atanf(PSIG * (patch - center));
                }
                pk.h[j] = (_Float16)comb;
            }
        }
        *(uint4*)(&sB[row * LDSTR + ks]) = pk.q;
        __syncthreads();

        // ---- fragment loads (per ISA 16-bit A/B VGPR layouts) ----
        const int koffA = (lane < 16) ? 0 : 8;      // lanes 0-15: K0-7/16-23
        const int koffB = (lane < 16) ? 0 : 8;      // lanes 0-15: K0-7/16-23
        Frag16 bf[2];
        #pragma unroll
        for (int cc = 0; cc < 2; ++cc) {
            const int nr = wn * 32 + cc * 16 + (lane & 15);
            bf[cc].q[0] = *(const uint4*)(&sB[nr * LDSTR + koffB]);
            bf[cc].q[1] = *(const uint4*)(&sB[nr * LDSTR + koffB + 16]);
        }
        #pragma unroll
        for (int rr = 0; rr < 3; ++rr) {
            Frag16 af;
            const int mr = wm * 48 + rr * 16 + (lane & 15);
            af.q[0] = *(const uint4*)(&sA[mr * LDSTR + koffA]);
            af.q[1] = *(const uint4*)(&sA[mr * LDSTR + koffA + 16]);
            #pragma unroll
            for (int cc = 0; cc < 2; ++cc) {
                acc[rr][cc] = __builtin_amdgcn_wmma_f32_16x16x32_f16(
                    false, af.v, false, bf[cc].v, (short)0, acc[rr][cc], false, false);
            }
        }
    }

    // ---- epilogue: +bias, relu, scatter to [B,192,H,W] ----
    #pragma unroll
    for (int cc = 0; cc < 2; ++cc) {
        const int nn  = n0 + wn * 32 + cc * 16 + (lane & 15);
        const int ob  = nn / HW;
        const int ohw = nn - ob * HW;
        float* op = out + (size_t)ob * 192 * HW + ohw;
        #pragma unroll
        for (int rr = 0; rr < 3; ++rr) {
            const int mbase = wm * 48 + rr * 16 + ((lane >> 4) << 3);
            #pragma unroll
            for (int r = 0; r < 8; ++r) {
                const int o = mbase + r;
                const float v = acc[rr][cc][r] + bias[o];
                op[(size_t)o * HW] = v > 0.0f ? v : 0.0f;
            }
        }
    }
}

// ---------------------------------------------------------------------------
// BN statistics: deterministic fixed-order tree reduction, one block/channel.
// stats[c] = sum, stats[192+c] = sum of squares, over [B, HW].
// ---------------------------------------------------------------------------
__global__ __launch_bounds__(256)
void bn_stats(const float* __restrict__ r, int B, int HW, float* __restrict__ stats) {
    const int c = blockIdx.x;
    float s = 0.0f, s2 = 0.0f;
    for (int b = 0; b < B; ++b) {
        const float* rp = r + ((size_t)b * 192 + c) * HW;
        for (int p = threadIdx.x; p < HW; p += 256) {
            float v = rp[p];
            s += v; s2 += v * v;
        }
    }
    __shared__ float sh[512];
    sh[threadIdx.x] = s;
    sh[256 + threadIdx.x] = s2;
    __syncthreads();
    for (int st = 128; st > 0; st >>= 1) {
        if (threadIdx.x < st) {
            sh[threadIdx.x]       += sh[threadIdx.x + st];
            sh[256 + threadIdx.x] += sh[256 + threadIdx.x + st];
        }
        __syncthreads();
    }
    if (threadIdx.x == 0) { stats[c] = sh[0]; stats[192 + c] = sh[256]; }
}

// ---------------------------------------------------------------------------
// BN + 2x2 maxpool: r[B,C,2Ho,2Wo] -> xo[B,C,Ho,Wo] (NCHW, d_out)
// and optionally xot[B,Ho*Wo,C] (NHWC staging for the next INRF layer).
// ---------------------------------------------------------------------------
__global__ void bn_norm_maxpool(const float* __restrict__ r, const float* __restrict__ stats,
                                const float* __restrict__ g, const float* __restrict__ be,
                                float* __restrict__ xo, float* __restrict__ xot,
                                int C, int Ho, int Wo, float invcnt) {
    int idx = blockIdx.x * 256 + threadIdx.x;
    int wo = idx % Wo; int t = idx / Wo;
    int ho = t % Ho;   t /= Ho;
    int c  = t % C;    int b = t / C;
    float mean = stats[c] * invcnt;
    float var  = stats[C + c] * invcnt - mean * mean;
    float ga = g[c] * rsqrtf(var + BN_EPS);
    float bb = be[c] - ga * mean;
    int Wi = Wo * 2;
    const float* rp = r + (((size_t)(b * C + c) * (Ho * 2) + ho * 2) * Wi + wo * 2);
    float m = fmaxf(fmaxf(ga * rp[0] + bb, ga * rp[1] + bb),
                    fmaxf(ga * rp[Wi] + bb, ga * rp[Wi + 1] + bb));
    xo[idx] = m;
    if (xot)
        xot[((size_t)b * (Ho * Wo) + ho * Wo + wo) * C + c] = m;
}

// ---------------------------------------------------------------------------
// BN in place on x01 (NCHW, d_out) + 2x2 maxpool into x03 stored NHWC
// ([B, Ho*Wo, C]) for the layer-3 implicit GEMM.
// ---------------------------------------------------------------------------
__global__ void bn_norm_inplace_pool(float* __restrict__ x01, const float* __restrict__ stats,
                                     const float* __restrict__ g, const float* __restrict__ be,
                                     float* __restrict__ x03t, int C, int Ho, int Wo, float invcnt) {
    int idx = blockIdx.x * 256 + threadIdx.x;
    int wo = idx % Wo; int t = idx / Wo;
    int ho = t % Ho;   t /= Ho;
    int c  = t % C;    int b = t / C;
    float mean = stats[c] * invcnt;
    float var  = stats[C + c] * invcnt - mean * mean;
    float ga = g[c] * rsqrtf(var + BN_EPS);
    float bb = be[c] - ga * mean;
    int Wi = Wo * 2;
    float* p = x01 + (((size_t)(b * C + c) * (Ho * 2) + ho * 2) * Wi + wo * 2);
    float v00 = ga * p[0] + bb;
    float v01 = ga * p[1] + bb;
    float v10 = ga * p[Wi] + bb;
    float v11 = ga * p[Wi + 1] + bb;
    p[0] = v00; p[1] = v01; p[Wi] = v10; p[Wi + 1] = v11;
    x03t[((size_t)b * (Ho * Wo) + ho * Wo + wo) * C + c] =
        fmaxf(fmaxf(v00, v01), fmaxf(v10, v11));
}

// ---------------------------------------------------------------------------
// BN + 8x8 avgpool: r3[B,192,64] -> z[B,192]; one 64-thread block per (b,c).
// ---------------------------------------------------------------------------
__global__ __launch_bounds__(64)
void bn_norm_avgpool(const float* __restrict__ r3, const float* __restrict__ stats,
                     const float* __restrict__ g, const float* __restrict__ be,
                     float* __restrict__ z, float invcnt) {
    const int bc = blockIdx.x;
    const int c  = bc % 192;
    float mean = stats[c] * invcnt;
    float var  = stats[192 + c] * invcnt - mean * mean;
    float ga = g[c] * rsqrtf(var + BN_EPS);
    float bb = be[c] - ga * mean;
    float v = ga * r3[(size_t)bc * 64 + threadIdx.x] + bb;
    __shared__ float sh[64];
    sh[threadIdx.x] = v;
    __syncthreads();
    for (int st = 32; st > 0; st >>= 1) {
        if (threadIdx.x < st) sh[threadIdx.x] += sh[threadIdx.x + st];
        __syncthreads();
    }
    if (threadIdx.x == 0) z[bc] = sh[0] * (1.0f / 64.0f);
}

// ---------------------------------------------------------------------------
// Final FC: logits[b][j] = z[b,:] . fcw[j,:] + fcb[j]
// ---------------------------------------------------------------------------
__global__ __launch_bounds__(128)
void fc_kernel(const float* __restrict__ z, const float* __restrict__ fcw,
               const float* __restrict__ fcb, float* __restrict__ logits) {
    const int b = blockIdx.x;
    const int j = threadIdx.x;
    if (j < 100) {
        float s = fcb[j];
        const float* zb = z + (size_t)b * 192;
        const float* wj = fcw + (size_t)j * 192;
        #pragma unroll 4
        for (int c = 0; c < 192; ++c) s += zb[c] * wj[c];
        logits[(size_t)b * 100 + j] = s;
    }
}

// ---------------------------------------------------------------------------
extern "C" void kernel_launch(void* const* d_in, const int* in_sizes, int n_in,
                              void* d_out, int out_size, void* d_ws, size_t ws_size,
                              hipStream_t stream) {
    const float* x   = (const float*)d_in[0];
    const float* w1  = (const float*)d_in[1];
    const float* b1  = (const float*)d_in[2];
    const float* w2  = (const float*)d_in[3];
    const float* b2  = (const float*)d_in[4];
    const float* w3  = (const float*)d_in[5];
    const float* b3  = (const float*)d_in[6];
    const float* g1  = (const float*)d_in[7];
    const float* be1 = (const float*)d_in[8];
    const float* g2  = (const float*)d_in[9];
    const float* be2 = (const float*)d_in[10];
    const float* g3  = (const float*)d_in[11];
    const float* be3 = (const float*)d_in[12];
    const float* fcw = (const float*)d_in[13];
    const float* fcb = (const float*)d_in[14];

    constexpr int B = 128;
    constexpr int KP1 = 96;    // ceil(75/32)*32
    constexpr int KP2 = 4800;  // 192*25
    // d_out layout: logits[12800] | x00[B*192*16*16] | x01[B*192*16*16]
    float* logits = (float*)d_out;
    float* x00    = logits + 12800;
    float* x01    = x00 + (size_t)B * 192 * 16 * 16;

    // workspace layout (bytes)
    char* ws = (char*)d_ws;
    _Float16* wpk1  = (_Float16*)(ws);                        // 192*96*2
    _Float16* wpk2  = (_Float16*)(ws + 36864);                // 192*4800*2
    _Float16* wpk3  = (_Float16*)(ws + 1880064);              // 192*4800*2
    float*    stats = (float*)(ws + 3723264);                 // 384 f32 (reused)
    float*    r1    = (float*)(ws + 3724800);                 // B*192*1024 f32 (100.7MB)
    // r1 region reused after layer1 (r1 dead once x00/x00t are produced):
    float*    x03t  = r1;                                     // B*64*192 f32 (NHWC)
    float*    r3    = (float*)((char*)r1 + (8u << 20));       // B*192*64 f32 (NCHW)
    float*    z     = (float*)((char*)r1 + (16u << 20));      // B*192 f32
    // optional NHWC staging of x00 (must NOT overlap r1; gate on ws_size)
    const size_t x00t_off  = 3724800ULL + 100663296ULL;
    const size_t need_nhwc = x00t_off + (size_t)B * 256 * 192 * 4;
    const bool   use_nhwc2 = (ws_size >= need_nhwc);
    float* x00t = (float*)(ws + x00t_off);                    // B*256*192 f32

    // ---- pack weights to f16 (K-major, zero padded) ----
    pack_weights<<<(192 * KP1 + 255) / 256, 256, 0, stream>>>(w1, wpk1, 3, KP1);
    pack_weights<<<(192 * KP2 + 255) / 256, 256, 0, stream>>>(w2, wpk2, 192, KP2);
    pack_weights<<<(192 * KP2 + 255) / 256, 256, 0, stream>>>(w3, wpk3, 192, KP2);

    // ---- layer 1: INRF(3->192, 32x32) + relu -> r1 ----
    inrf_gemm<3, 32, 32, KP1, 75, false><<<B * 1024 / 64, 256, 0, stream>>>(x, wpk1, b1, r1);
    bn_stats<<<192, 256, 0, stream>>>(r1, B, 1024, stats);
    bn_norm_maxpool<<<(B * 192 * 16 * 16) / 256, 256, 0, stream>>>(
        r1, stats, g1, be1, x00, use_nhwc2 ? x00t : nullptr, 192, 16, 16,
        1.0f / (B * 1024.0f));

    // ---- layer 2: INRF(192->192, 16x16) + relu -> x01 (raw), BN inplace + pool ----
    if (use_nhwc2)
        inrf_gemm<192, 16, 16, KP2, KP2, true><<<B * 256 / 64, 256, 0, stream>>>(
            x00t, wpk2, b2, x01);
    else
        inrf_gemm<192, 16, 16, KP2, KP2, false><<<B * 256 / 64, 256, 0, stream>>>(
            x00, wpk2, b2, x01);
    bn_stats<<<192, 256, 0, stream>>>(x01, B, 256, stats);
    bn_norm_inplace_pool<<<(B * 192 * 8 * 8) / 256, 256, 0, stream>>>(
        x01, stats, g2, be2, x03t, 192, 8, 8, 1.0f / (B * 256.0f));

    // ---- layer 3: INRF(192->192, 8x8) + relu -> r3, BN + avgpool -> z ----
    inrf_gemm<192, 8, 8, KP2, KP2, true><<<B * 64 / 64, 256, 0, stream>>>(x03t, wpk3, b3, r3);
    bn_stats<<<192, 256, 0, stream>>>(r3, B, 64, stats);
    bn_norm_avgpool<<<B * 192, 64, 0, stream>>>(r3, stats, g3, be3, z, 1.0f / (B * 64.0f));

    // ---- FC head ----
    fc_kernel<<<B, 128, 0, stream>>>(z, fcw, fcb, logits);
}
